// Up_sampling_75213467287640
// MI455X (gfx1250) — matrix operand order repaired and checked by reading, not compiled, not run on hardware
//
#include <hip/hip_runtime.h>

typedef __attribute__((ext_vector_type(16))) _Float16 v16h;
typedef __attribute__((ext_vector_type(8)))  float    v8f;

// ---------------------------------------------------------------------------
// 3-NN inverse-distance interpolation.
// One thread per query; reference points tiled through LDS.
// ---------------------------------------------------------------------------
#define KNN_TPB 256
#define KNN_TILE 256

__global__ void __launch_bounds__(KNN_TPB)
knn_interp_kernel(const float* __restrict__ pq, const float* __restrict__ pr,
                  const float* __restrict__ xr, float* __restrict__ out,
                  int Nq, int Nr, int Cr)
{
    __shared__ float sp[KNN_TILE * 3];
    const int q = blockIdx.x * KNN_TPB + threadIdx.x;
    float qx = 0.f, qy = 0.f, qz = 0.f;
    if (q < Nq) { qx = pq[q*3+0]; qy = pq[q*3+1]; qz = pq[q*3+2]; }

    float b0 = 1e30f, b1 = 1e30f, b2 = 1e30f;
    int   i0 = 0, i1 = 0, i2 = 0;

    for (int t0 = 0; t0 < Nr; t0 += KNN_TILE) {
        int cnt = Nr - t0; if (cnt > KNN_TILE) cnt = KNN_TILE;
        __syncthreads();
        for (int j = threadIdx.x; j < cnt * 3; j += KNN_TPB)
            sp[j] = pr[t0 * 3 + j];
        __syncthreads();
        for (int j = 0; j < cnt; ++j) {
            float dx = qx - sp[j*3+0];
            float dy = qy - sp[j*3+1];
            float dz = qz - sp[j*3+2];
            float d2 = dx*dx + dy*dy + dz*dz;
            int ri = t0 + j;
            if (d2 < b0)      { b2=b1; i2=i1; b1=b0; i1=i0; b0=d2; i0=ri; }
            else if (d2 < b1) { b2=b1; i2=i1; b1=d2; i1=ri; }
            else if (d2 < b2) { b2=d2; i2=ri; }
        }
    }
    if (q >= Nq) return;

    float w0 = 1.f / (fmaxf(b0, 0.f) + 1e-8f);
    float w1 = 1.f / (fmaxf(b1, 0.f) + 1e-8f);
    float w2 = 1.f / (fmaxf(b2, 0.f) + 1e-8f);
    float ws = w0 + w1 + w2;
    w0 /= ws; w1 /= ws; w2 /= ws;

    const float* r0 = xr + (long)i0 * Cr;
    const float* r1 = xr + (long)i1 * Cr;
    const float* r2 = xr + (long)i2 * Cr;
    float* o = out + (long)q * Cr;
    for (int c = 0; c < Cr; ++c)
        o[c] = w0 * r0[c] + w1 * r1[c] + w2 * r2[c];
}

// ---------------------------------------------------------------------------
// Y(N x Cout) = X(N x K) * W(Cout x K)^T + b via v_wmma_f32_16x16x32_f16.
// Register-blocked: each wave owns a 64x16 output strip (4 M-tiles x 1 N-tile)
// so one B (weight) fragment feeds 4 WMMAs per K-step.
// Requires N%64==0, K%32==0, Cout%16==0 (true for all levels).
// ---------------------------------------------------------------------------
__global__ void __launch_bounds__(128)
gemm_bias_wmma(const float* __restrict__ X, const float* __restrict__ W,
               const float* __restrict__ bias, float* __restrict__ Y,
               int N, int K, int Cout)
{
    const int lane    = threadIdx.x & 31;
    const int wave    = threadIdx.x >> 5;
    const int tiles_n = Cout >> 4;
    const int total   = (N >> 6) * tiles_n;      // 64-row M blocks
    const int t       = blockIdx.x * 4 + wave;
    if (t >= total) return;                      // wave-uniform; EXEC stays all-1s

    const int  mb    = t / tiles_n;              // 64-row block index
    const int  nt    = t % tiles_n;
    const int  half  = lane >> 4;                // 0: lanes 0-15, 1: lanes 16-31
    const int  l15   = lane & 15;
    const int  bcol  = (nt << 4) + l15;          // B column == W row for this lane
    const long arow0 = (long)(mb << 6) + l15;    // first A row for this lane

    v8f acc[4] = {};
    for (int k0 = 0; k0 < K; k0 += 32) {
        // B 32x16 f16 (B = W^T): lane<16 -> K 0..15, lane>=16 -> K 16..31
        v16h b;
        const float* bp = W + (long)bcol * K + k0 + half * 16;
        #pragma unroll
        for (int j = 0; j < 16; ++j)
            b[j] = (_Float16)bp[j];

        // A 16x32 f16: lane<16 -> K 0..15, lane>=16 -> K 16..31
        v16h a[4];
        #pragma unroll
        for (int m = 0; m < 4; ++m) {
            const float* ap = X + (arow0 + m * 16) * K + k0 + half * 16;
            #pragma unroll
            for (int j = 0; j < 8; ++j) {
                a[m][j]     = (_Float16)ap[j];
                a[m][8 + j] = (_Float16)ap[8 + j];
            }
        }

        #pragma unroll
        for (int m = 0; m < 4; ++m)
            acc[m] = __builtin_amdgcn_wmma_f32_16x16x32_f16(
                         /*neg_a=*/false, a[m], /*neg_b=*/false, b,
                         /*c_mod=*/(short)0, acc[m],
                         /*reuse_a=*/false, /*reuse_b=*/false);
    }

    // D layout: lane -> N = nt*16 + l15 ; VGPR j -> M = mtile*16 + half*8 + j
    const float bv = bias[bcol];
    #pragma unroll
    for (int m = 0; m < 4; ++m) {
        const long mrow0 = (long)(mb << 6) + m * 16 + half * 8;
        #pragma unroll
        for (int j = 0; j < 8; ++j)
            Y[(mrow0 + j) * Cout + bcol] = acc[m][j] + bv;
    }
}

// ---------------------------------------------------------------------------
// Batch-stat computation: mean and rstd per channel (biased variance).
// One block per channel.
// ---------------------------------------------------------------------------
__global__ void __launch_bounds__(256)
bn_stats_kernel(const float* __restrict__ t, int N, int C,
                float* __restrict__ mean, float* __restrict__ rstd)
{
    __shared__ float s1[256], s2[256];
    const int c = blockIdx.x;
    float a = 0.f, b = 0.f;
    for (int i = threadIdx.x; i < N; i += 256) {
        float v = t[(long)i * C + c];
        a += v; b += v * v;
    }
    s1[threadIdx.x] = a; s2[threadIdx.x] = b;
    __syncthreads();
    for (int off = 128; off > 0; off >>= 1) {
        if ((int)threadIdx.x < off) {
            s1[threadIdx.x] += s1[threadIdx.x + off];
            s2[threadIdx.x] += s2[threadIdx.x + off];
        }
        __syncthreads();
    }
    if (threadIdx.x == 0) {
        float m   = s1[0] / (float)N;
        float var = s2[0] / (float)N - m * m;
        mean[c] = m;
        rstd[c] = rsqrtf(var + 1e-5f);
    }
}

// ---------------------------------------------------------------------------
// y = relu((t - mean) * rstd * g + beta) [+ residual]
// ---------------------------------------------------------------------------
__global__ void __launch_bounds__(256)
bn_apply_kernel(const float* __restrict__ t, const float* __restrict__ mean,
                const float* __restrict__ rstd, const float* __restrict__ g,
                const float* __restrict__ beta, const float* __restrict__ residual,
                float* __restrict__ out, int N, int C)
{
    long i = (long)blockIdx.x * 256 + threadIdx.x;
    if (i >= (long)N * C) return;
    int c = (int)(i % C);
    float v = (t[i] - mean[c]) * rstd[c] * g[c] + beta[c];
    v = fmaxf(v, 0.f);
    if (residual) v += residual[i];
    out[i] = v;
}

// ---------------------------------------------------------------------------
// Host orchestration
// ---------------------------------------------------------------------------
extern "C" void kernel_launch(void* const* d_in, const int* in_sizes, int n_in,
                              void* d_out, int out_size, void* d_ws, size_t ws_size,
                              hipStream_t stream)
{
    (void)in_sizes; (void)n_in; (void)out_size; (void)ws_size;

    static const int  Ns[5]     = {16384, 4096, 1024, 256, 64};
    static const int  Cs[5]     = {32, 64, 128, 256, 512};
    static const long outOff[4] = {0, 524288, 786432, 917504}; // y1..y4

    float* ws         = (float*)d_ws;
    float* buf_interp = ws;                    // 16384*64 = 1,048,576 floats (max Nq*2C)
    float* buf_t      = ws + 1048576;          //   524,288 floats (max Nq*C)
    float* buf_x      = buf_interp;            // reuse: interp dead once first GEMM ran
    float* mean       = ws + 1048576 + 524288; // 256 floats
    float* rstd       = mean + 256;            // 256 floats

    for (int l = 1; l <= 4; ++l) {
        const int Nq = Ns[l-1], Nr = Ns[l];
        const int C  = Cs[l-1], Cr = Cs[l];          // Cr == 2*C
        const float* pq = (const float*)d_in[l-1];
        const float* pr = (const float*)d_in[l];
        const float* xq = (const float*)d_in[5 + l - 1];   // skip connection
        const float* xr = (const float*)d_in[5 + l];       // coarse features
        const int wb = 10 + 8 * (4 - l);                   // weights for this level
        const float* Wc = (const float*)d_in[wb+0];
        const float* bc = (const float*)d_in[wb+1];
        const float* gc = (const float*)d_in[wb+2];
        const float* Bc = (const float*)d_in[wb+3];
        const float* Wo = (const float*)d_in[wb+4];
        const float* bo = (const float*)d_in[wb+5];
        const float* go = (const float*)d_in[wb+6];
        const float* Bo = (const float*)d_in[wb+7];
        float* yout = (float*)d_out + outOff[l-1];

        // 1) 3-NN inverse-distance interpolation -> (Nq, 2C)
        knn_interp_kernel<<<Nq / KNN_TPB, KNN_TPB, 0, stream>>>(
            pq, pr, xr, buf_interp, Nq, Nr, Cr);

        // 2) Linear (2C -> C) + bias
        {
            int wtiles = (Nq >> 6) * (C >> 4);
            gemm_bias_wmma<<<(wtiles + 3) / 4, 128, 0, stream>>>(
                buf_interp, Wc, bc, buf_t, Nq, Cr, C);
        }
        // 3) BN + ReLU + residual add -> x
        bn_stats_kernel<<<C, 256, 0, stream>>>(buf_t, Nq, C, mean, rstd);
        bn_apply_kernel<<<((long)Nq * C + 255) / 256, 256, 0, stream>>>(
            buf_t, mean, rstd, gc, Bc, xq, buf_x, Nq, C);

        // 4) Linear (C -> C) + bias
        {
            int wtiles = (Nq >> 6) * (C >> 4);
            gemm_bias_wmma<<<(wtiles + 3) / 4, 128, 0, stream>>>(
                buf_x, Wo, bo, buf_t, Nq, C, C);
        }
        // 5) BN + ReLU -> output
        bn_stats_kernel<<<C, 256, 0, stream>>>(buf_t, Nq, C, mean, rstd);
        bn_apply_kernel<<<((long)Nq * C + 255) / 256, 256, 0, stream>>>(
            buf_t, mean, rstd, go, Bo, nullptr, yout, Nq, C);
    }

    // passthrough: x4 -> tail of output (64*512 floats at offset 983040)
    hipMemcpyAsync((float*)d_out + 983040, d_in[9], 32768 * sizeof(float),
                   hipMemcpyDeviceToDevice, stream);
}
